// GMSIFN_70987219468901
// MI455X (gfx1250) — compile-verified, matrix operand order checked by reading, not run on hardware
//
#include <hip/hip_runtime.h>
#include <cstdint>
#include <cstddef>

// ---------------------------------------------------------------------------
// Problem constants (match reference)
// ---------------------------------------------------------------------------
#define DD        240
#define FIN       5120
#define OUTD      16
#define BB        16
#define NNODE     512
#define TOPK      5
#define BN        (BB * NNODE)        // 8192 rows
#define NBR_ROWS  (BN * TOPK)         // 40960 rows
#define CATLD     256                 // padded ld for concat(nbr,bond) buffer

typedef __attribute__((ext_vector_type(16))) __bf16 v16bf;
typedef __attribute__((ext_vector_type(8)))  float  v8f;
typedef __attribute__((ext_vector_type(4)))  float  f32x4;

// ---------------------------------------------------------------------------
// Device helpers
// ---------------------------------------------------------------------------
__device__ __forceinline__ float lrelu(float x) { return x > 0.f ? x : 0.01f * x; }
__device__ __forceinline__ float sigm(float x)  { return 1.f / (1.f + __expf(-x)); }

// Full (unguarded) bf16 fragment load from f32 row: two contiguous 8-elem runs.
// Matches the documented 16-bit A/B VGPR layout: lane r holds the row; hi half
// selects K offsets {hi*8 .. hi*8+7} and {16+hi*8 .. 16+hi*8+7} in a K=32 step.
__device__ __forceinline__ v16bf frag_full(const float* __restrict__ r0, int hi) {
  const f32x4* pa = (const f32x4*)(r0 + hi * 8);
  const f32x4* pb = (const f32x4*)(r0 + 16 + hi * 8);
  f32x4 a0 = pa[0], a1 = pa[1];
  f32x4 b0 = pb[0], b1 = pb[1];
  v16bf f;
#pragma unroll
  for (int i = 0; i < 4; ++i) {
    f[i]      = (__bf16)a0[i];
    f[4 + i]  = (__bf16)a1[i];
    f[8 + i]  = (__bf16)b0[i];
    f[12 + i] = (__bf16)b1[i];
  }
  return f;
}

// Guarded tail fragment (K not multiple of 32): zero-fill out-of-range K.
__device__ __forceinline__ v16bf frag_tail(const float* __restrict__ r0, int K, int k0, int hi) {
  v16bf f;
#pragma unroll
  for (int i = 0; i < 8; ++i) {
    int ka = k0 + hi * 8 + i;
    int kb = k0 + 16 + hi * 8 + i;
    f[i]     = (ka < K) ? (__bf16)r0[ka] : (__bf16)0.f;
    f[8 + i] = (kb < K) ? (__bf16)r0[kb] : (__bf16)0.f;
  }
  return f;
}

#define WMMA_BF16(a, b, c) \
  __builtin_amdgcn_wmma_f32_16x16x32_bf16(false, (a), false, (b), (short)0, (c), false, false)

// ---------------------------------------------------------------------------
// Generic WMMA GEMM:  C[M,N] = act(scale * (A[M,K] @ W[N,K]^T) + bias[N])
// A, W row-major f32 (K contiguous), converted to bf16 in-register.
// Batched via blockIdx.z with element strides. 128 threads = 4 waves; each
// wave owns a 32x32 output tile (2 A-frags x 2 B-frags -> 4 WMMAs per K step,
// doubling math-per-byte vs a 16x16 tile). Workgroup covers 64x64.
// tm/tn guards (wave-uniform) handle M or N edges that are only 16 wide.
// act: 0 = identity, 1 = leaky_relu(0.01)
// ---------------------------------------------------------------------------
__global__ void __launch_bounds__(128)
k_gemm(const float* __restrict__ A, long long sAb, int lda,
       const float* __restrict__ W, long long sWb, int ldw,
       const float* __restrict__ bias,
       float* __restrict__ C, long long sCb, int ldc,
       int M, int N, int K, float scale, int act)
{
  A += (long long)blockIdx.z * sAb;
  W += (long long)blockIdx.z * sWb;
  C += (long long)blockIdx.z * sCb;

  const int wave = threadIdx.x >> 5;
  const int lane = threadIdx.x & 31;
  const int r    = lane & 15;
  const int hi   = lane >> 4;
  const int m0   = blockIdx.y * 64 + (wave >> 1) * 32;
  const int n0   = blockIdx.x * 64 + (wave & 1) * 32;
  if (m0 >= M || n0 >= N) return;   // whole-wave uniform exit; EXEC stays full
  const bool tm = (m0 + 16) < M;    // second M sub-tile exists
  const bool tn = (n0 + 16) < N;    // second N sub-tile exists

  const float* rA0 = A + (long long)(m0 + r) * lda;
  const float* rA1 = A + (long long)(m0 + 16 + r) * lda;
  const float* rW0 = W + (long long)(n0 + r) * ldw;
  const float* rW1 = W + (long long)(n0 + 16 + r) * ldw;

  v8f acc00 = {}, acc01 = {}, acc10 = {}, acc11 = {};
  const int kmain = K & ~31;
  for (int k0 = 0; k0 < kmain; k0 += 32) {
    if (k0 + 32 < K) {
      __builtin_prefetch(rA0 + k0 + 32, 0, 1);   // global_prefetch_b8
      __builtin_prefetch(rW0 + k0 + 32, 0, 1);
      if (tm) __builtin_prefetch(rA1 + k0 + 32, 0, 1);
      if (tn) __builtin_prefetch(rW1 + k0 + 32, 0, 1);
    }
    v16bf a0 = frag_full(rA0 + k0, hi);
    v16bf b0 = frag_full(rW0 + k0, hi);
    v16bf a1 = {};
    acc00 = WMMA_BF16(a0, b0, acc00);
    if (tm) { a1 = frag_full(rA1 + k0, hi); acc10 = WMMA_BF16(a1, b0, acc10); }
    if (tn) {
      v16bf b1 = frag_full(rW1 + k0, hi);
      acc01 = WMMA_BF16(a0, b1, acc01);
      if (tm) acc11 = WMMA_BF16(a1, b1, acc11);
    }
  }
  if (kmain < K) {
    v16bf a0 = frag_tail(rA0, K, kmain, hi);
    v16bf b0 = frag_tail(rW0, K, kmain, hi);
    v16bf a1 = {};
    acc00 = WMMA_BF16(a0, b0, acc00);
    if (tm) { a1 = frag_tail(rA1, K, kmain, hi); acc10 = WMMA_BF16(a1, b0, acc10); }
    if (tn) {
      v16bf b1 = frag_tail(rW1, K, kmain, hi);
      acc01 = WMMA_BF16(a0, b1, acc01);
      if (tm) acc11 = WMMA_BF16(a1, b1, acc11);
    }
  }

  // C/D layout: VGPR g -> row g + 8*hi, col = lane & 15
  const int c0 = n0 + r;
  const int c1 = n0 + 16 + r;
  const float bv0 = bias ? bias[c0] : 0.f;
  const float bv1 = (bias && tn) ? bias[c1] : 0.f;
#pragma unroll
  for (int g = 0; g < 8; ++g) {
    const long long row0 = (long long)(m0 + hi * 8 + g) * ldc;
    float v = acc00[g] * scale + bv0;
    if (act == 1) v = lrelu(v);
    C[row0 + c0] = v;
    if (tn) {
      float u = acc01[g] * scale + bv1;
      if (act == 1) u = lrelu(u);
      C[row0 + c1] = u;
    }
    if (tm) {
      const long long row1 = (long long)(m0 + 16 + hi * 8 + g) * ldc;
      float w = acc10[g] * scale + bv0;
      if (act == 1) w = lrelu(w);
      C[row1 + c0] = w;
      if (tn) {
        float z = acc11[g] * scale + bv1;
        if (act == 1) z = lrelu(z);
        C[row1 + c1] = z;
      }
    }
  }
}

// ---------------------------------------------------------------------------
// Elementwise / small kernels
// ---------------------------------------------------------------------------
__global__ void k_hv(const float* __restrict__ h, const float* __restrict__ v,
                     float* __restrict__ hv, long long rows, int D_) {
  long long i = (long long)blockIdx.x * blockDim.x + threadIdx.x;
  if (i >= rows * D_) return;
  hv[i] = h[i] * v[(int)(i % D_)];
}

__global__ void k_relu(const float* __restrict__ s, float* __restrict__ d, long long n) {
  long long i = (long long)blockIdx.x * blockDim.x + threadIdx.x;
  if (i >= n) return;
  d[i] = fmaxf(s[i], 0.f);
}

__global__ void k_pad_copy(const float* __restrict__ src, float* __restrict__ dst,
                           int rows, int K, int ldd) {
  long long i = (long long)blockIdx.x * blockDim.x + threadIdx.x;
  if (i >= (long long)rows * ldd) return;
  int c = (int)(i % ldd);
  long long rI = i / ldd;
  dst[i] = (c < K) ? src[rI * K + c] : 0.f;
}

// top-k (k=5, descending, first-occurrence ties) per row of sim
__global__ void k_topk(const float* __restrict__ sim, float* __restrict__ vals,
                       int* __restrict__ idx, int rows, int n) {
  int i = blockIdx.x * blockDim.x + threadIdx.x;
  if (i >= rows) return;
  const float* row = sim + (long long)i * n;
  float tv[TOPK]; int ti[TOPK];
#pragma unroll
  for (int k = 0; k < TOPK; ++k) { tv[k] = -3.4e38f; ti[k] = 0; }
  for (int m = 0; m < n; ++m) {
    float s = row[m];
    if (s > tv[TOPK - 1]) {
      int p = TOPK - 1;
      while (p > 0 && tv[p - 1] < s) { tv[p] = tv[p - 1]; ti[p] = ti[p - 1]; --p; }
      tv[p] = s; ti[p] = m;
    }
  }
#pragma unroll
  for (int k = 0; k < TOPK; ++k) {
    vals[(long long)i * TOPK + k] = tv[k];
    idx [(long long)i * TOPK + k] = ti[k];
  }
}

// Gather neighbor features; optionally append bond = sigmoid(simval) at col D.
__global__ void k_gather(const float* __restrict__ src, const int* __restrict__ idx,
                         const float* __restrict__ vals, float* __restrict__ dst,
                         long long nTotal /* B*N*K rows */, int Kk, int N_, int D_,
                         int ldd, int has_bond) {
  long long i = (long long)blockIdx.x * blockDim.x + threadIdx.x;
  if (i >= nTotal * ldd) return;
  int c = (int)(i % ldd);
  long long tK = i / ldd;
  float v = 0.f;
  if (c < D_) {
    long long bn = tK / Kk;
    long long b  = bn / N_;
    int j = idx[tK];
    v = src[(b * N_ + j) * (long long)D_ + c];
  } else if (c == D_ && has_bond) {
    v = sigm(vals[tK]);
  }
  dst[i] = v;
}

// GAT-style attention over K=5 neighbors. One wave per (b,n) row.
// score_k = leaky(dot(node_act, aW[0:D]) + dot(nbrF_k, aW[D:2D]) + ab)
// ctx = elu( sum_k softmax(score)_k * t_k )
__global__ void k_nbr_attn(const float* __restrict__ node_act, const float* __restrict__ nbrF,
                           const float* __restrict__ t, const float* __restrict__ aW,
                           const float* __restrict__ ab, float* __restrict__ ctx,
                           int rows, int D_) {
  int wid  = blockIdx.x * (blockDim.x >> 5) + (threadIdx.x >> 5);
  int lane = threadIdx.x & 31;
  if (wid >= rows) return;

  float sA = 0.f;
  for (int d = lane; d < D_; d += 32) sA += node_act[(long long)wid * D_ + d] * aW[d];
  for (int o = 16; o; o >>= 1) sA += __shfl_xor(sA, o, 32);

  float w[TOPK];
  float mx = -3.4e38f;
#pragma unroll
  for (int k = 0; k < TOPK; ++k) {
    const float* nr = nbrF + ((long long)wid * TOPK + k) * D_;
    float s = 0.f;
    for (int d = lane; d < D_; d += 32) s += nr[d] * aW[D_ + d];
    for (int o = 16; o; o >>= 1) s += __shfl_xor(s, o, 32);
    s = lrelu(sA + s + ab[0]);
    w[k] = s; mx = fmaxf(mx, s);
  }
  float den = 0.f;
#pragma unroll
  for (int k = 0; k < TOPK; ++k) { w[k] = __expf(w[k] - mx); den += w[k]; }
  float inv = 1.f / den;
  for (int d = lane; d < D_; d += 32) {
    float a = 0.f;
#pragma unroll
    for (int k = 0; k < TOPK; ++k) a += w[k] * t[((long long)wid * TOPK + k) * D_ + d];
    a *= inv;
    ctx[(long long)wid * D_ + d] = a > 0.f ? a : (__expf(a) - 1.f);
  }
}

// Fused GRU gate combine: gi/gh are [rows, 3D] pre-activations, h is [rows, D].
__global__ void k_gru(const float* __restrict__ gi, const float* __restrict__ gh,
                      const float* __restrict__ h, float* __restrict__ out,
                      long long rows, int D_) {
  long long i = (long long)blockIdx.x * blockDim.x + threadIdx.x;
  if (i >= rows * D_) return;
  int c = (int)(i % D_);
  long long rI = i / D_;
  const float* gir = gi + rI * (long long)(3 * D_);
  const float* ghr = gh + rI * (long long)(3 * D_);
  float rg = sigm(gir[c] + ghr[c]);
  float zg = sigm(gir[D_ + c] + ghr[D_ + c]);
  float ng = tanhf(gir[2 * D_ + c] + rg * ghr[2 * D_ + c]);
  out[i] = (1.f - zg) * ng + zg * h[i];
}

// graph_feature = sum over nodes; also keeps a pristine copy + relu
__global__ void k_graph_sum(const float* __restrict__ act, float* __restrict__ gfeat,
                            float* __restrict__ gfeat2, float* __restrict__ gact,
                            int B_, int N_, int D_) {
  int i = blockIdx.x * blockDim.x + threadIdx.x;
  if (i >= B_ * D_) return;
  int b = i / D_, d = i % D_;
  float s = 0.f;
  for (int n = 0; n < N_; ++n) s += act[((long long)b * N_ + n) * D_ + d];
  gfeat[i] = s; gfeat2[i] = s; gact[i] = fmaxf(s, 0.f);
}

// Graph-level attention: one 256-thread block per batch. flip reads act/t at B-1-b.
__global__ void __launch_bounds__(256)
k_graph_attn(const float* __restrict__ gact, const float* __restrict__ act,
             const float* __restrict__ t, const float* __restrict__ aW,
             const float* __restrict__ ab, float* __restrict__ gctx,
             int B_, int N_, int D_, int flip) {
  __shared__ float sc[NNODE];
  __shared__ float red[256];
  const int b   = blockIdx.x;
  const int bs  = flip ? (B_ - 1 - b) : b;
  const int tid = threadIdx.x;

  float p = 0.f;
  for (int d = tid; d < D_; d += 256) p += gact[(long long)b * D_ + d] * aW[d];
  red[tid] = p; __syncthreads();
  for (int s = 128; s > 0; s >>= 1) { if (tid < s) red[tid] += red[tid + s]; __syncthreads(); }
  const float sA = red[0]; __syncthreads();

  for (int n = tid; n < N_; n += 256) {
    const float* ar = act + ((long long)bs * N_ + n) * D_;
    float s = 0.f;
    for (int d = 0; d < D_; ++d) s += ar[d] * aW[D_ + d];
    sc[n] = lrelu(sA + s + ab[0]);
  }
  __syncthreads();

  float m = -3.4e38f;
  for (int n = tid; n < N_; n += 256) m = fmaxf(m, sc[n]);
  red[tid] = m; __syncthreads();
  for (int s = 128; s > 0; s >>= 1) { if (tid < s) red[tid] = fmaxf(red[tid], red[tid + s]); __syncthreads(); }
  const float mx = red[0]; __syncthreads();

  float e = 0.f;
  for (int n = tid; n < N_; n += 256) { float v = __expf(sc[n] - mx); sc[n] = v; e += v; }
  red[tid] = e; __syncthreads();
  for (int s = 128; s > 0; s >>= 1) { if (tid < s) red[tid] += red[tid + s]; __syncthreads(); }
  const float den = red[0]; __syncthreads();

  const float inv = 1.f / den;
  for (int d = tid; d < D_; d += 256) {
    float a = 0.f;
    for (int n = 0; n < N_; ++n) a += sc[n] * t[((long long)bs * N_ + n) * D_ + d];
    a *= inv;
    gctx[(long long)b * D_ + d] = a > 0.f ? a : (__expf(a) - 1.f);
  }
}

__global__ void k_flip_relu(const float* __restrict__ src, float* __restrict__ dstFlip,
                            float* __restrict__ dstAct, int B_, int D_) {
  int i = blockIdx.x * blockDim.x + threadIdx.x;
  if (i >= B_ * D_) return;
  int b = i / D_, d = i % D_;
  float v = src[(B_ - 1 - b) * D_ + d];
  dstFlip[i] = v;
  dstAct[i]  = fmaxf(v, 0.f);
}

__global__ void k_fused(const float* __restrict__ a, const float* __restrict__ brev,
                        float* __restrict__ o, int B_, int D_) {
  int i = blockIdx.x * blockDim.x + threadIdx.x;
  if (i >= B_ * D_) return;
  int b = i / D_, d = i % D_;
  o[i] = a[i] + brev[(B_ - 1 - b) * D_ + d];
}

// ---------------------------------------------------------------------------
// Host driver
// ---------------------------------------------------------------------------
extern "C" void kernel_launch(void* const* d_in, const int* in_sizes, int n_in,
                              void* d_out, int out_size, void* d_ws, size_t ws_size,
                              hipStream_t stream) {
  (void)in_sizes; (void)n_in; (void)out_size; (void)ws_size;
  const float* input = (const float*)d_in[0];
  auto P = [&](int i) -> const float* { return (const float*)d_in[1 + i]; };

  // Bump allocator over workspace (256B aligned, 16B load-alignment guaranteed).
  char* base = (char*)d_ws;
  size_t off = 0;
  auto allocf = [&](size_t nElems) -> float* {
    float* p = (float*)(base + off);
    off += ((nElems * sizeof(float) + 255) / 256) * 256;
    return p;
  };

  float* xf    = allocf((size_t)BN * DD);
  float* hf    = allocf((size_t)BN * DD);
  float* hv    = allocf((size_t)BN * DD);
  float* sim   = allocf((size_t)BB * NNODE * NNODE);
  float* tvals = allocf((size_t)BN * TOPK);
  int*   tidx  = (int*)allocf((size_t)BN * TOPK);
  float* catf  = allocf((size_t)NBR_ROWS * CATLD);
  float* p2pad = allocf((size_t)DD * CATLD);
  float* nf0   = allocf((size_t)BN * DD);
  float* nbrf  = allocf((size_t)NBR_ROWS * DD);
  float* tbuf  = allocf((size_t)NBR_ROWS * DD);
  float* ctx   = allocf((size_t)BN * DD);
  float* gi    = allocf((size_t)BN * 3 * DD);
  float* gh    = allocf((size_t)BN * 3 * DD);
  float* hsA   = allocf((size_t)BN * DD);
  float* actA  = allocf((size_t)BN * DD);
  float* hsB   = allocf((size_t)BN * DD);
  float* actB  = allocf((size_t)BN * DD);
  float* tg1   = allocf((size_t)BN * DD);
  float* tg2   = allocf((size_t)BN * DD);
  float* gfeat = allocf((size_t)BB * DD);
  float* gact  = allocf((size_t)BB * DD);
  float* gfin  = allocf((size_t)BB * DD);
  float* gctx  = allocf((size_t)BB * DD);
  float* gig   = allocf((size_t)BB * 3 * DD);
  float* ghg   = allocf((size_t)BB * 3 * DD);
  float* gfr   = allocf((size_t)BB * DD);
  float* gactr = allocf((size_t)BB * DD);
  float* fused = allocf((size_t)BB * DD);

  auto cdiv = [](long long a, long long b) -> unsigned { return (unsigned)((a + b - 1) / b); };
  auto gemm = [&](const float* A, long long sAb, int lda,
                  const float* W, long long sWb, int ldw,
                  const float* bias, float* C, long long sCb, int ldc,
                  int M, int N, int K, float scale, int act, int batches) {
    dim3 g(cdiv(N, 64), cdiv(M, 64), (unsigned)batches);
    k_gemm<<<g, 128, 0, stream>>>(A, sAb, lda, W, sWb, ldw, bias, C, sCb, ldc,
                                  M, N, K, scale, act);
  };
  auto ewg = [&](long long n) { return dim3(cdiv(n, 256)); };

  const float SIM_SCALE = 0.064549722f; // 1/sqrt(240)

  // ---- x = leaky(input @ p38^T + p39)  [8192, 240] ----
  gemm(input, 0, FIN, P(38), 0, FIN, P(39), xf, 0, DD, BN, DD, FIN, 1.f, 1, 1);

  // ---- GGL round 1 on x ----
  gemm(xf, 0, DD, P(40), 0, DD, P(41), hf, 0, DD, BN, DD, DD, 1.f, 1, 1);
  k_hv<<<ewg((long long)BN * DD), 256, 0, stream>>>(hf, P(42), hv, BN, DD);
  gemm(hv, (long long)NNODE * DD, DD, hf, (long long)NNODE * DD, DD, nullptr,
       sim, (long long)NNODE * NNODE, NNODE, NNODE, NNODE, DD, SIM_SCALE, 0, BB);
  k_topk<<<ewg(BN), 256, 0, stream>>>(sim, tvals, tidx, BN, NNODE);
  k_gather<<<ewg((long long)NBR_ROWS * CATLD), 256, 0, stream>>>(
      xf, tidx, tvals, catf, NBR_ROWS, TOPK, NNODE, DD, CATLD, 1);

  // ---- node_fc, neighbor_fc ----
  gemm(xf, 0, DD, P(0), 0, DD, P(1), nf0, 0, DD, BN, DD, DD, 1.f, 1, 1);
  k_pad_copy<<<ewg((long long)DD * CATLD), 256, 0, stream>>>(P(2), p2pad, DD, DD + 1, CATLD);
  gemm(catf, 0, CATLD, p2pad, 0, CATLD, P(3), nbrf, 0, DD,
       NBR_ROWS, DD, CATLD, 1.f, 1, 1);

  // ---- neighbor attention round 1 + GRU1 ----
  gemm(nbrf, 0, DD, P(16), 0, DD, P(17), tbuf, 0, DD, NBR_ROWS, DD, DD, 1.f, 0, 1);
  k_nbr_attn<<<cdiv((long long)BN * 32, 256), 256, 0, stream>>>(
      nf0, nbrf, tbuf, P(12), P(13), ctx, BN, DD);
  gemm(ctx, 0, DD, P(4), 0, DD, P(6), gi, 0, 3 * DD, BN, 3 * DD, DD, 1.f, 0, 1);
  gemm(nf0, 0, DD, P(5), 0, DD, P(7), gh, 0, 3 * DD, BN, 3 * DD, DD, 1.f, 0, 1);
  k_gru<<<ewg((long long)BN * DD), 256, 0, stream>>>(gi, gh, nf0, hsA, BN, DD);
  k_relu<<<ewg((long long)BN * DD), 256, 0, stream>>>(hsA, actA, (long long)BN * DD);

  // ---- round 2 (RADIUS-1): GGL on act, attention, GRU2 ----
  gemm(actA, 0, DD, P(40), 0, DD, P(41), hf, 0, DD, BN, DD, DD, 1.f, 1, 1);
  k_hv<<<ewg((long long)BN * DD), 256, 0, stream>>>(hf, P(42), hv, BN, DD);
  gemm(hv, (long long)NNODE * DD, DD, hf, (long long)NNODE * DD, DD, nullptr,
       sim, (long long)NNODE * NNODE, NNODE, NNODE, NNODE, DD, SIM_SCALE, 0, BB);
  k_topk<<<ewg(BN), 256, 0, stream>>>(sim, tvals, tidx, BN, NNODE);
  k_gather<<<ewg((long long)NBR_ROWS * DD), 256, 0, stream>>>(
      actA, tidx, tvals, catf, NBR_ROWS, TOPK, NNODE, DD, DD, 0);  // raw nbr2, ld=D
  gemm(catf, 0, DD, P(18), 0, DD, P(19), tbuf, 0, DD, NBR_ROWS, DD, DD, 1.f, 0, 1);
  k_nbr_attn<<<cdiv((long long)BN * 32, 256), 256, 0, stream>>>(
      actA, catf, tbuf, P(14), P(15), ctx, BN, DD);
  gemm(ctx, 0, DD, P(8), 0, DD, P(10), gi, 0, 3 * DD, BN, 3 * DD, DD, 1.f, 0, 1);
  gemm(hsA, 0, DD, P(9), 0, DD, P(11), gh, 0, 3 * DD, BN, 3 * DD, DD, 1.f, 0, 1);
  k_gru<<<ewg((long long)BN * DD), 256, 0, stream>>>(gi, gh, hsA, hsB, BN, DD);
  k_relu<<<ewg((long long)BN * DD), 256, 0, stream>>>(hsB, actB, (long long)BN * DD);

  // ---- graph feature + precomputed attend transforms ----
  k_graph_sum<<<ewg(BB * DD), 256, 0, stream>>>(actB, gfeat, gfin, gact, BB, NNODE, DD);
  gemm(actB, 0, DD, P(26), 0, DD, P(27), tg1, 0, DD, BN, DD, DD, 1.f, 0, 1);
  gemm(actB, 0, DD, P(34), 0, DD, P(35), tg2, 0, DD, BN, DD, DD, 1.f, 0, 1);

  // ---- graph GRU loop (forward), T_ITERS = 2 ----
  for (int it = 0; it < 2; ++it) {
    k_graph_attn<<<BB, 256, 0, stream>>>(gact, actB, tg1, P(24), P(25), gctx,
                                         BB, NNODE, DD, 0);
    gemm(gctx, 0, DD, P(20), 0, DD, P(22), gig, 0, 3 * DD, BB, 3 * DD, DD, 1.f, 0, 1);
    gemm(gfeat, 0, DD, P(21), 0, DD, P(23), ghg, 0, 3 * DD, BB, 3 * DD, DD, 1.f, 0, 1);
    k_gru<<<ewg(BB * DD), 256, 0, stream>>>(gig, ghg, gfeat, gfeat, BB, DD);
    k_relu<<<ewg(BB * DD), 256, 0, stream>>>(gfeat, gact, (long long)BB * DD);
  }

  // ---- graph GRU loop (batch-reversed), T_ITERS = 2 ----
  k_flip_relu<<<ewg(BB * DD), 256, 0, stream>>>(gfin, gfr, gactr, BB, DD);
  for (int it = 0; it < 2; ++it) {
    k_graph_attn<<<BB, 256, 0, stream>>>(gactr, actB, tg2, P(32), P(33), gctx,
                                         BB, NNODE, DD, 1);
    gemm(gctx, 0, DD, P(28), 0, DD, P(30), gig, 0, 3 * DD, BB, 3 * DD, DD, 1.f, 0, 1);
    gemm(gfr, 0, DD, P(29), 0, DD, P(31), ghg, 0, 3 * DD, BB, 3 * DD, DD, 1.f, 0, 1);
    k_gru<<<ewg(BB * DD), 256, 0, stream>>>(gig, ghg, gfr, gfr, BB, DD);
    k_relu<<<ewg(BB * DD), 256, 0, stream>>>(gfr, gactr, (long long)BB * DD);
  }

  // ---- fuse + output projection straight into d_out ----
  k_fused<<<ewg(BB * DD), 256, 0, stream>>>(gfeat, gfr, fused, BB, DD);
  gemm(fused, 0, DD, P(36), 0, DD, P(37), (float*)d_out, 0, OUTD,
       BB, OUTD, DD, 1.f, 0, 1);
}
